// HardAttention_39170101739770
// MI455X (gfx1250) — matrix operand order repaired and checked by reading, not compiled, not run on hardware
//
#include <hip/hip_runtime.h>
#include <hip/hip_bf16.h>
#include <math.h>

#define B_ 128
#define N_ 4096
#define D_ 256
#define H_ 512
#define U_ 256

#define ROWS 128          // N-rows per workgroup in the score kernel
#define PADK (D_ + 16)    // padded LDS row (bf16 elems) to spread banks
#define SPITCH 132        // scratch pitch (floats): half-waves 32 banks apart

typedef __attribute__((ext_vector_type(16))) __bf16 v16bf;
typedef __attribute__((ext_vector_type(8)))  float  v8f;

union Frag { v16bf v; uint4 q[2]; };

static __device__ __forceinline__ unsigned short f2bf(float f) {
  union { float f; unsigned u; } c; c.f = f;
  unsigned r = c.u + 0x7FFFu + ((c.u >> 16) & 1u);   // round-to-nearest-even
  return (unsigned short)(r >> 16);
}

static __device__ __forceinline__ float fast_tanh(float x) {
#if __has_builtin(__builtin_amdgcn_tanhf)
  return __builtin_amdgcn_tanhf(x);       // V_TANH_F32 on gfx1250
#else
  return tanhf(x);
#endif
}

// ---------------- kernel 1: hsum[b][u] = hidden@W2 + W2_b + W1_b ----------------
__global__ __launch_bounds__(U_)
void k_hproj(const float* __restrict__ hidden, const float* __restrict__ W2w,
             const float* __restrict__ W2b, const float* __restrict__ W1b,
             float* __restrict__ hsum) {
  const int b = blockIdx.x;
  const int u = threadIdx.x;
  float acc = W2b[u] + W1b[u];
  const float* hrow = hidden + b * H_;
#pragma unroll 4
  for (int k = 0; k < H_; ++k)
    acc = fmaf(hrow[k], W2w[k * U_ + u], acc);
  hsum[b * U_ + u] = acc;
}

// ---------------- kernel 2: W1 (D x U, f32) -> W1^T (U x D, bf16) ----------------
__global__ __launch_bounds__(256)
void k_w1t(const float* __restrict__ W1w, unsigned short* __restrict__ w1t) {
  int idx = blockIdx.x * 256 + threadIdx.x;  // 65536 elements
  int k = idx & (D_ - 1);
  int u = idx >> 8;
  w1t[u * D_ + k] = f2bf(W1w[k * U_ + u]);
}

// ---------------- kernel 3: fused scores GEMM (bf16 WMMA) + tanh + dot(V) -------
// 256 threads = 8 waves; request 4 waves/SIMD so 2 WGs co-reside per WGP and the
// global->LDS staging phase of one WG overlaps the WMMA phase of the other.
__global__ __launch_bounds__(256, 4)
void k_scores(const float* __restrict__ features,
              const unsigned short* __restrict__ w1t,
              const float* __restrict__ hsum,
              const float* __restrict__ Vw, const float* __restrict__ Vb,
              float* __restrict__ logits) {
  __shared__ unsigned short feat[ROWS * PADK];   // bf16 tile, padded rows (68 KB)
  __shared__ float scratch[ROWS * SPITCH];       // per-column logit partials (66 KB)

  const int b    = blockIdx.x >> 5;            // / (N_/ROWS)
  const int n0   = (blockIdx.x & 31) * ROWS;
  const int tid  = threadIdx.x;
  const int wid  = tid >> 5;                   // wave 0..7
  const int lane = tid & 31;
  const int l16  = lane & 15;
  const bool hih = lane >= 16;                 // high half-wave

  // cooperative load: 128x256 f32 -> bf16 LDS
  const float* fsrc = features + ((size_t)b * N_ + n0) * D_;
#pragma unroll 4
  for (int it = 0; it < (ROWS * D_ / 4) / 256; ++it) {
    int v   = it * 256 + tid;                  // float4 index (linear over tile)
    int row = v >> 6;                          // (v*4)/256
    int col = (v & 63) * 4;
    float4 f = ((const float4*)fsrc)[v];
    unsigned lo = (unsigned)f2bf(f.x) | ((unsigned)f2bf(f.y) << 16);
    unsigned hi = (unsigned)f2bf(f.z) | ((unsigned)f2bf(f.w) << 16);
    unsigned* p = (unsigned*)&feat[row * PADK + col];
    p[0] = lo; p[1] = hi;
  }

  // persistent B fragments: this wave owns u-tiles [2*wid, 2*wid+1], all 8 K-steps
  Frag bf[2][8];
  const int ut0 = wid * 2;
#pragma unroll
  for (int j = 0; j < 2; ++j) {
    const unsigned short* wp = w1t + ((ut0 + j) * 16 + l16) * D_;
#pragma unroll
    for (int s = 0; s < 8; ++s) {
      int k0 = s * 32 + (hih ? 16 : 0);        // lanes 0-15: K 0..15, 16-31: K 16..31
      bf[j][s].q[0] = *(const uint4*)(wp + k0);
      bf[j][s].q[1] = *(const uint4*)(wp + k0 + 8);
    }
  }
  const float hs0 = hsum[b * U_ + ut0 * 16 + l16];
  const float hs1 = hsum[b * U_ + (ut0 + 1) * 16 + l16];
  const float vw0 = Vw[ut0 * 16 + l16];
  const float vw1 = Vw[(ut0 + 1) * 16 + l16];

  __syncthreads();

  // per-lane base: A layout (16-bit, 16x32): lane = row, halves split K by 16 (contiguous)
  const unsigned short* ap0 = &feat[l16 * PADK + (hih ? 16 : 0)];
  // per-lane scratch slot: column index = wid*16 + l16; row base offset for half
  float* sp0 = &scratch[(hih ? 8 : 0) * SPITCH + wid * 16 + l16];

#pragma unroll
  for (int m = 0; m < ROWS / 16; ++m) {
    // batched A-fragment loads for this 16-row tile (16x ds_load_b128 as a clause)
    Frag af[8];
#pragma unroll
    for (int s = 0; s < 8; ++s) {
      const unsigned short* p = ap0 + m * (16 * PADK) + s * 32;
      af[s].q[0] = *(const uint4*)(p);
      af[s].q[1] = *(const uint4*)(p + 8);
    }
    __builtin_amdgcn_sched_barrier(0);         // keep loads batched ahead of WMMAs

    v8f acc0 = {};
    v8f acc1 = {};
#pragma unroll
    for (int s = 0; s < 8; ++s) {
      acc0 = __builtin_amdgcn_wmma_f32_16x16x32_bf16(false, af[s].v, false,
                                                     bf[0][s].v, (short)0, acc0,
                                                     false, false);
      acc1 = __builtin_amdgcn_wmma_f32_16x16x32_bf16(false, af[s].v, false,
                                                     bf[1][s].v, (short)0, acc1,
                                                     false, false);
    }

    // epilogue: tanh + V-weighting; fire-and-forget per-column partial stores
    float* sp = sp0 + m * (16 * SPITCH);
#pragma unroll
    for (int r = 0; r < 8; ++r) {
      float p = fast_tanh(acc0[r] + hs0) * vw0 + fast_tanh(acc1[r] + hs1) * vw1;
      sp[r * SPITCH] = p;                      // ds_store_b32, no wait needed here
    }
  }

  __syncthreads();

  // final: one thread per row sums its 128 column partials (pipelined b128 loads)
  if (tid < ROWS) {
    const float* rp = &scratch[tid * SPITCH];
    float s = 0.0f;
#pragma unroll
    for (int j = 0; j < 128; j += 4) {
      float4 v = *(const float4*)(rp + j);
      s += (v.x + v.y) + (v.z + v.w);
    }
    logits[(size_t)b * N_ + n0 + tid] = s + Vb[0];
  }
}

// ---------------- kernel 4: softmax over N, argmax, gather ----------------------
__global__ __launch_bounds__(256)
void k_softmax(const float* __restrict__ logits, const float* __restrict__ features,
               float* __restrict__ out_ctx, float* __restrict__ out_aw,
               float* __restrict__ out_idx) {
  __shared__ float smax[256];
  __shared__ int   sidx[256];
  __shared__ float ssum[256];
  const int b = blockIdx.x;
  const int t = threadIdx.x;
  const float* l = logits + (size_t)b * N_;

  float bv = -INFINITY; int bi = 0;
  for (int i = t; i < N_; i += 256) {          // ascending -> first occurrence kept
    float v = l[i];
    if (v > bv) { bv = v; bi = i; }
  }
  smax[t] = bv; sidx[t] = bi;
  __syncthreads();
  for (int off = 128; off > 0; off >>= 1) {
    if (t < off) {
      float ov = smax[t + off]; int oi = sidx[t + off];
      if (ov > smax[t] || (ov == smax[t] && oi < sidx[t])) { smax[t] = ov; sidx[t] = oi; }
    }
    __syncthreads();
  }
  const float gmax = smax[0];
  const int   gidx = sidx[0];

  float s = 0.0f;
  for (int i = t; i < N_; i += 256) {
    float e = __expf(l[i] - gmax);
    out_aw[(size_t)b * N_ + i] = e;
    s += e;
  }
  ssum[t] = s;
  __syncthreads();
  for (int off = 128; off > 0; off >>= 1) {
    if (t < off) ssum[t] += ssum[t + off];
    __syncthreads();
  }
  const float inv = 1.0f / ssum[0];
  for (int i = t; i < N_; i += 256)
    out_aw[(size_t)b * N_ + i] *= inv;

  out_ctx[b * D_ + t] = features[((size_t)b * N_ + gidx) * D_ + t];
  if (t == 0) out_idx[b] = (float)gidx;
}

// ---------------- launcher ------------------------------------------------------
extern "C" void kernel_launch(void* const* d_in, const int* in_sizes, int n_in,
                              void* d_out, int out_size, void* d_ws, size_t ws_size,
                              hipStream_t stream) {
  const float* features = (const float*)d_in[0];
  const float* hidden   = (const float*)d_in[1];
  const float* W1w      = (const float*)d_in[2];
  const float* W1b      = (const float*)d_in[3];
  const float* W2w      = (const float*)d_in[4];
  const float* W2b      = (const float*)d_in[5];
  const float* Vw       = (const float*)d_in[6];
  const float* Vb       = (const float*)d_in[7];

  float*          ws_hsum   = (float*)d_ws;                                   // 128 KB
  unsigned short* ws_w1t    = (unsigned short*)((char*)d_ws + B_ * U_ * 4);   // 128 KB
  float*          ws_logits = (float*)((char*)d_ws + B_ * U_ * 4 + D_ * U_ * 2); // 2 MB

  float* out_ctx = (float*)d_out;                 // (B, D)
  float* out_aw  = out_ctx + B_ * D_;             // (B, N, 1)
  float* out_idx = out_aw + (size_t)B_ * N_;      // (B,) as float

  k_hproj  <<<B_, U_, 0, stream>>>(hidden, W2w, W2b, W1b, ws_hsum);
  k_w1t    <<<(D_ * U_) / 256, 256, 0, stream>>>(W1w, ws_w1t);
  k_scores <<<B_ * (N_ / ROWS), 256, 0, stream>>>(features, ws_w1t, ws_hsum, Vw, Vb, ws_logits);
  k_softmax<<<B_, 256, 0, stream>>>(ws_logits, features, out_ctx, out_aw, out_idx);
}
